// MiMoV2MoEGate_36490042147118
// MI455X (gfx1250) — matrix operand order repaired and checked by reading, not compiled, not run on hardware
//
#include <hip/hip_runtime.h>
#include <hip/hip_bf16.h>
#include <math.h>

typedef __attribute__((ext_vector_type(16))) __bf16        v16bf;
typedef __attribute__((ext_vector_type(8)))  float         v8f;
typedef __attribute__((ext_vector_type(4)))  unsigned int  u32x4;
typedef __attribute__((ext_vector_type(4)))  float         f32x4;

#define H          4096
#define N_EXPERTS  256
#define TOP_K      8
#define N_GROUP    8
#define GROUP_SZ   32
#define TOPK_GROUP 4
#define SCALING    2.5f

#define KCHUNK   128
#define NCHUNK   (H / KCHUNK)          // 32
#define TOK_BLK  32                    // tokens per block
#define XSTRIDE  136                   // padded bf16 row stride (bank advance 4/row)
#define SSTRIDE  264                   // padded float row stride for score tile

__device__ __forceinline__ unsigned short f2bf(float f) {
  unsigned int u = __float_as_uint(f);
  u += 0x7FFFu + ((u >> 16) & 1u);     // round-to-nearest-even
  return (unsigned short)(u >> 16);
}

// One-time weight swizzle: fp32 [256][4096] -> bf16 in WMMA B-fragment order.
// wsz[((et*128 + ks)*32 + lane)*16 + j]:
//   expert = et*16 + (lane&15),  k = ks*32 + ((lane&16)?8:0) + (j<8 ? j : j+8)
__global__ __launch_bounds__(256)
void prep_w_kernel(const float* __restrict__ w, unsigned short* __restrict__ wsz) {
  int gid  = blockIdx.x * blockDim.x + threadIdx.x;   // 65536 total
  int lane = gid & 31;
  int ks   = (gid >> 5) & 127;
  int et   = gid >> 12;                               // 0..15
  int e    = et * 16 + (lane & 15);
  int kb   = ks * 32 + ((lane & 16) ? 8 : 0);
  const float* src = w + (size_t)e * H + kb;
  unsigned short tmp[16];
#pragma unroll
  for (int j = 0; j < 16; ++j) tmp[j] = f2bf(src[(j < 8) ? j : j + 8]);
  u32x4* dst = (u32x4*)(wsz + (size_t)gid * 16);
  dst[0] = *(const u32x4*)(tmp);
  dst[1] = *(const u32x4*)(tmp + 8);
}

typedef union { u32x4 q[2]; v16bf v; } Frag;

__device__ __forceinline__ float sigm(float v) {
  return 1.0f / (1.0f + __expf(-v));
}

__global__ __launch_bounds__(256)
void gate_kernel(const float* __restrict__ x,
                 const unsigned short* __restrict__ wsz,
                 const float* __restrict__ bias,
                 int* __restrict__ out_idx,
                 float* __restrict__ out_w) {
  __shared__ unsigned short xt[2][TOK_BLK * XSTRIDE]; // double-buffered bf16 x tile
  __shared__ float          sc[TOK_BLK * SSTRIDE];    // sigmoid scores [32 tok][256 exp]
  __shared__ float          bs[N_EXPERTS];

  const int tid     = threadIdx.x;
  const int lane    = tid & 31;
  const int wid     = tid >> 5;
  const int tokBase = blockIdx.x * TOK_BLK;

  bs[tid] = bias[tid];                         // 256 threads == 256 experts

  const int et0 = wid * 2, et1 = et0 + 1;      // two 16-expert tiles per wave
  v8f c00 = {}, c01 = {}, c10 = {}, c11 = {};  // [token-tile][expert-tile]

  // cooperative x staging: thread -> (row, 16 contiguous floats = 64B)
  const int xrow = tid >> 3;                   // 0..31
  const int xcol = (tid & 7) << 4;             // 0..112
  const float* xsrc    = x + (size_t)(tokBase + xrow) * H + xcol;
  const int    xdstOff = xrow * XSTRIDE + xcol;

  // A-fragment addressing (ISA 16-bit A 16x32 layout)
  const int aM   = lane & 15;
  const int aoff = (lane < 16) ? 0 : 8;

  // B-fragment bases (pre-swizzled; 32B contiguous per lane)
  const unsigned short* b0base = wsz + ((size_t)et0 * 128 * 32 + lane) * 16;
  const unsigned short* b1base = wsz + ((size_t)et1 * 128 * 32 + lane) * 16;

  // ---- prologue: stage chunk 0 into buffer 0 ----
  {
    f32x4 r0 = *(const f32x4*)(xsrc);
    f32x4 r1 = *(const f32x4*)(xsrc + 4);
    f32x4 r2 = *(const f32x4*)(xsrc + 8);
    f32x4 r3 = *(const f32x4*)(xsrc + 12);
    unsigned short hb[16];
#pragma unroll
    for (int j = 0; j < 4; ++j) {
      hb[j]      = f2bf(r0[j]); hb[4 + j]  = f2bf(r1[j]);
      hb[8 + j]  = f2bf(r2[j]); hb[12 + j] = f2bf(r3[j]);
    }
    u32x4* d = (u32x4*)(&xt[0][xdstOff]);
    d[0] = *(const u32x4*)(hb);
    d[1] = *(const u32x4*)(hb + 8);
  }
  __syncthreads();

  for (int c = 0; c < NCHUNK; ++c) {
    const int cur  = c & 1;
    const int nxtb = cur ^ 1;
    const int cn   = (c + 1 < NCHUNK) ? (c + 1) : c;
    const float* snext = xsrc + cn * KCHUNK;

    // start next-chunk x moving NOW via prefetch: global_prefetch does NOT
    // touch LOADcnt, so it cannot serialize with the B-fragment waits below.
    __builtin_prefetch(snext, 0, 3);

    const unsigned short* a0b = &xt[cur][aM * XSTRIDE];
    const unsigned short* a1b = &xt[cur][(16 + aM) * XSTRIDE];
    const int ksb = c * (KCHUNK / 32);

    // software-pipelined 4 k-steps: preload step i+1 while WMMA'ing step i
    Frag A0[2], A1[2], B0[2], B1[2];
    A0[0].q[0] = *(const u32x4*)(a0b + aoff);
    A0[0].q[1] = *(const u32x4*)(a0b + 16 + aoff);
    A1[0].q[0] = *(const u32x4*)(a1b + aoff);
    A1[0].q[1] = *(const u32x4*)(a1b + 16 + aoff);
    {
      const u32x4* p0 = (const u32x4*)(b0base + (size_t)ksb * 512);
      const u32x4* p1 = (const u32x4*)(b1base + (size_t)ksb * 512);
      B0[0].q[0] = p0[0]; B0[0].q[1] = p0[1];
      B1[0].q[0] = p1[0]; B1[0].q[1] = p1[1];
    }
#pragma unroll
    for (int i = 0; i < KCHUNK / 32; ++i) {
      const int pc = i & 1, pn = pc ^ 1;
      if (i < KCHUNK / 32 - 1) {
        A0[pn].q[0] = *(const u32x4*)(a0b + (i + 1) * 32 + aoff);
        A0[pn].q[1] = *(const u32x4*)(a0b + (i + 1) * 32 + 16 + aoff);
        A1[pn].q[0] = *(const u32x4*)(a1b + (i + 1) * 32 + aoff);
        A1[pn].q[1] = *(const u32x4*)(a1b + (i + 1) * 32 + 16 + aoff);
        const u32x4* p0 = (const u32x4*)(b0base + (size_t)(ksb + i + 1) * 512);
        const u32x4* p1 = (const u32x4*)(b1base + (size_t)(ksb + i + 1) * 512);
        B0[pn].q[0] = p0[0]; B0[pn].q[1] = p0[1];
        B1[pn].q[0] = p1[0]; B1[pn].q[1] = p1[1];
      }
      c00 = __builtin_amdgcn_wmma_f32_16x16x32_bf16(false, A0[pc].v, false, B0[pc].v,
                                                    (short)0, c00, false, false);
      c01 = __builtin_amdgcn_wmma_f32_16x16x32_bf16(false, A0[pc].v, false, B1[pc].v,
                                                    (short)0, c01, false, false);
      c10 = __builtin_amdgcn_wmma_f32_16x16x32_bf16(false, A1[pc].v, false, B0[pc].v,
                                                    (short)0, c10, false, false);
      c11 = __builtin_amdgcn_wmma_f32_16x16x32_bf16(false, A1[pc].v, false, B1[pc].v,
                                                    (short)0, c11, false, false);
    }

    // Scheduling fence: keep the real x loads BELOW every B load of this
    // chunk, so in-order LOADcnt retirement never makes a B wait include them.
    asm volatile("" ::: "memory");

    // real x loads: issued last, hit prefetched lines, waited only here
    {
      f32x4 r0 = *(const f32x4*)(snext);
      f32x4 r1 = *(const f32x4*)(snext + 4);
      f32x4 r2 = *(const f32x4*)(snext + 8);
      f32x4 r3 = *(const f32x4*)(snext + 12);
      unsigned short hb[16];
#pragma unroll
      for (int j = 0; j < 4; ++j) {
        hb[j]      = f2bf(r0[j]); hb[4 + j]  = f2bf(r1[j]);
        hb[8 + j]  = f2bf(r2[j]); hb[12 + j] = f2bf(r3[j]);
      }
      u32x4* d = (u32x4*)(&xt[nxtb][xdstOff]);
      d[0] = *(const u32x4*)(hb);
      d[1] = *(const u32x4*)(hb + 8);
    }
    __syncthreads();
  }

  // C/D layout: VGPR r, lanes 0-15 -> (M=r, N=lane), lanes 16-31 -> (M=r+8, N=lane-16)
  const int cN    = lane & 15;
  const int rbase = (lane < 16) ? 0 : 8;
#pragma unroll
  for (int r = 0; r < 8; ++r) {
    sc[(rbase + r) * SSTRIDE + et0 * 16 + cN]        = sigm(c00[r]);
    sc[(rbase + r) * SSTRIDE + et1 * 16 + cN]        = sigm(c01[r]);
    sc[(16 + rbase + r) * SSTRIDE + et0 * 16 + cN]   = sigm(c10[r]);
    sc[(16 + rbase + r) * SSTRIDE + et1 * 16 + cN]   = sigm(c11[r]);
  }
  __syncthreads();

  // ---- selection: 4 tokens per wave (lanes 0,8,16,24), 32 tokens in parallel ----
  if ((lane & 7) == 0) {
    const int tok = wid * 4 + (lane >> 3);
    const float* srow = sc + tok * SSTRIDE;

    // group score = sum of top-2 bias-corrected scores per group
    float gsum[N_GROUP];
#pragma unroll
    for (int g = 0; g < N_GROUP; ++g) {
      float m1 = -INFINITY, m2 = -INFINITY;
      for (int i = 0; i < GROUP_SZ; ++i) {
        float v = srow[g * GROUP_SZ + i] + bs[g * GROUP_SZ + i];
        if (v > m1) { m2 = m1; m1 = v; }
        else if (v > m2) { m2 = v; }
      }
      gsum[g] = m1 + m2;
    }
    // top-4 groups (strict > keeps lowest index on ties, matching lax.top_k)
    unsigned gm = 0;
    for (int r = 0; r < TOPK_GROUP; ++r) {
      float best = -INFINITY; int bg = 0;
#pragma unroll
      for (int g = 0; g < N_GROUP; ++g)
        if (!((gm >> g) & 1u) && gsum[g] > best) { best = gsum[g]; bg = g; }
      gm |= 1u << bg;
    }
    // stable top-8 over allowed groups (ascending e; ties keep earlier index)
    float val[TOP_K], raw[TOP_K]; int idx[TOP_K];
#pragma unroll
    for (int j = 0; j < TOP_K; ++j) { val[j] = -INFINITY; raw[j] = 0.f; idx[j] = 0; }
    for (int g = 0; g < N_GROUP; ++g) {
      if (!((gm >> g) & 1u)) continue;
      for (int i = 0; i < GROUP_SZ; ++i) {
        const int e = g * GROUP_SZ + i;
        const float rv = srow[e];
        const float v  = rv + bs[e];
        if (v > val[TOP_K - 1]) {
          float cv = v, cr = rv; int ce = e;
#pragma unroll
          for (int p = 0; p < TOP_K; ++p) {
            if (cv > val[p]) {
              float tv = val[p]; val[p] = cv; cv = tv;
              float tr = raw[p]; raw[p] = cr; cr = tr;
              int   te = idx[p]; idx[p] = ce; ce = te;
            }
          }
        }
      }
    }
    float denom = 1e-20f;
#pragma unroll
    for (int j = 0; j < TOP_K; ++j) denom += raw[j];
    const float sscale = SCALING / denom;
    const int t = tokBase + tok;
#pragma unroll
    for (int j = 0; j < TOP_K; ++j) {
      out_idx[t * TOP_K + j] = idx[j];
      out_w[t * TOP_K + j]   = raw[j] * sscale;
    }
  }
}

extern "C" void kernel_launch(void* const* d_in, const int* in_sizes, int n_in,
                              void* d_out, int out_size, void* d_ws, size_t ws_size,
                              hipStream_t stream) {
  (void)n_in; (void)out_size; (void)ws_size;
  const float* x    = (const float*)d_in[0];   // [4,4096,4096] fp32
  const float* w    = (const float*)d_in[1];   // [256,4096]    fp32
  const float* bias = (const float*)d_in[2];   // [256]         fp32
  const int T = in_sizes[0] / H;               // 16384 tokens

  unsigned short* wsz = (unsigned short*)d_ws; // 2 MB bf16 swizzled weights
  int*   out_idx = (int*)d_out;                        // first T*8 int32
  float* out_w   = (float*)d_out + (size_t)T * TOP_K;  // next  T*8 fp32

  prep_w_kernel<<<(N_EXPERTS / 16) * 128 * 32 / 256, 256, 0, stream>>>(w, wsz);
  gate_kernel<<<T / TOK_BLK, 256, 0, stream>>>(x, wsz, bias, out_idx, out_w);
}
